// MultiViewCrossAttentionVoxTransformerExtractor_54254026883693
// MI455X (gfx1250) — compile-verified, matrix operand order checked by reading, not run on hardware
//
#include <hip/hip_runtime.h>
#include <hip/hip_bf16.h>
#include <math.h>

typedef _Float16 f16;
typedef __attribute__((ext_vector_type(16))) _Float16 v16h;
typedef __attribute__((ext_vector_type(8)))  _Float16 v8h;
typedef __attribute__((ext_vector_type(8)))  float    v8f;

#define PE_ORDER   8
#define SP_DIM     (PE_ORDER*3)        // 24
#define NUM_HEADS  4
#define DIM_KEY    32
#define DIM_VALUE  32
#define M_VIEWS    8
#define F_DIM      32
#define AUG_DIM    (F_DIM+16)          // 48
#define HD         (NUM_HEADS*DIM_KEY) // 128
#define TILE_VOX   32
#define THREADS    512                 // 16 waves
#define NWAVE      16
#define NT         8                   // 128 cols / 16
#define ROWS       (TILE_VOX*M_VIEWS)  // 256 (voxel,view) rows

// fragment element counts (halves); layout [tiles][ksteps][lane(32)][elem(16)]
#define AUGF_N (16*2*32*16) // 16384
#define WKF_N  (8*2*32*16)  // 8192
#define WQF_N  (8*32*16)    // 4096
#define WVF_N  (8*32*16)    // 4096
#define ENCF_N (2*32*16)    // 1024 (two 16-row M-tiles)

#define SMEM_BYTES ((AUGF_N+WKF_N+WQF_N+WVF_N+ENCF_N)*2 /* frag halves */ \
                  + (ROWS*HD)*2*2                        /* sKh + sVh f16 */ \
                  + (TILE_VOX*HD + TILE_VOX*32 + TILE_VOX*8 + TILE_VOX*3)*4 /* floats */ \
                  + (TILE_VOX + TILE_VOX*M_VIEWS)*4)     /* ints */

__device__ __forceinline__ unsigned f32_to_ordered(float x) {
  unsigned b = __float_as_uint(x);
  return (b & 0x80000000u) ? ~b : (b | 0x80000000u);
}
__device__ __forceinline__ float ordered_to_f32(unsigned u) {
  unsigned b = (u & 0x80000000u) ? (u ^ 0x80000000u) : ~u;
  return __uint_as_float(b);
}

__global__ void init_minmax_kernel(unsigned* ws) {
  int t = threadIdx.x;
  if (t < 3) { ws[t] = 0xFFFFFFFFu; ws[3 + t] = 0u; }
}

__global__ void reduce_minmax_kernel(const float* __restrict__ voxels, int N, unsigned* ws) {
  unsigned mn[3] = {0xFFFFFFFFu, 0xFFFFFFFFu, 0xFFFFFFFFu};
  unsigned mx[3] = {0u, 0u, 0u};
  int stride = gridDim.x * blockDim.x;
  for (int i = blockIdx.x * blockDim.x + threadIdx.x; i < N; i += stride) {
#pragma unroll
    for (int a = 0; a < 3; a++) {
      unsigned u = f32_to_ordered(voxels[i * 3 + a]);
      mn[a] = (u < mn[a]) ? u : mn[a];
      mx[a] = (u > mx[a]) ? u : mx[a];
    }
  }
#pragma unroll
  for (int a = 0; a < 3; a++) {
    atomicMin(&ws[a], mn[a]);
    atomicMax(&ws[3 + a], mx[a]);
  }
}

__global__ __launch_bounds__(THREADS) void mvca_kernel(
    const int* __restrict__ vox_indices,
    const float* __restrict__ voxels,
    const float* __restrict__ voxel_features,
    const float* __restrict__ scores,
    const int* __restrict__ camera_ids,
    const float* __restrict__ extrinsics,
    const float* __restrict__ Wq,
    const float* __restrict__ Wk,
    const float* __restrict__ Wv,
    float* __restrict__ out,
    const unsigned* __restrict__ ws,
    int B, int numTiles) {
  extern __shared__ __align__(32) char smem[];
  f16* sAugF = (f16*)smem;              // [16 mtile][2 kstep][32 lane][16]
  f16* sWkF  = sAugF + AUGF_N;          // [8 ntile][2 kstep][32][16]
  f16* sWqF  = sWkF + WKF_N;            // [8 ntile][32][16]
  f16* sWvF  = sWqF + WQF_N;            // [8 ntile][32][16]
  f16* sEncF = sWvF + WVF_N;            // [2 qtile][32][16]
  f16* sKh   = sEncF + ENCF_N;          // [256][128] f16
  f16* sVh   = sKh + ROWS * HD;         // [256][128] f16
  float* sQ    = (float*)(sVh + ROWS * HD); // [32][128]
  float* sLog  = sQ + TILE_VOX * HD;        // [32][32] (h*8+m)
  float* sMask = sLog + TILE_VOX * 32;      // [32][8]
  float* sXYZ  = sMask + TILE_VOX * 8;      // [32][3]
  int* sVidx   = (int*)(sXYZ + TILE_VOX * 3); // [32]
  int* sCam    = sVidx + TILE_VOX;            // [32][8]

  const int tid  = threadIdx.x;
  const int lane = tid & 31;
  const int wave = tid >> 5;

  // ---- Stage weight fragments ONCE per (persistent) workgroup ----
  for (int i = tid; i < WQF_N; i += THREADS) {
    int nt = i >> 9; int L = (i >> 4) & 31; int e = i & 15;
    int n = nt * 16 + (L & 15);
    int k = ((e >> 3) << 4) + ((L >> 4) << 3) + (e & 7);
    sWqF[i] = (f16)((k < SP_DIM) ? Wq[k * HD + n] : 0.f);
  }
  for (int i = tid; i < WKF_N; i += THREADS) {
    int nt = i >> 10; int ks = (i >> 9) & 1; int L = (i >> 4) & 31; int e = i & 15;
    int n = nt * 16 + (L & 15);
    int k = ks * 32 + ((e >> 3) << 4) + ((L >> 4) << 3) + (e & 7);
    sWkF[i] = (f16)((k < AUG_DIM) ? Wk[k * HD + n] : 0.f);
  }
  for (int i = tid; i < WVF_N; i += THREADS) {
    int nt = i >> 9; int L = (i >> 4) & 31; int e = i & 15;
    int n = nt * 16 + (L & 15);
    int k = ((e >> 3) << 4) + ((L >> 4) << 3) + (e & 7);
    sWvF[i] = (f16)Wv[k * HD + n];
  }
  const float lo0 = ordered_to_f32(ws[0]), hi0 = ordered_to_f32(ws[3]);
  const float lo1 = ordered_to_f32(ws[1]), hi1 = ordered_to_f32(ws[4]);
  const float lo2 = ordered_to_f32(ws[2]), hi2 = ordered_to_f32(ws[5]);

  for (int tile = blockIdx.x; tile < numTiles; tile += gridDim.x) {
    __syncthreads();  // weights ready / previous iteration fully consumed

    // ---- Phase B: indices, cameras, mask, normalized coords ----
    if (tid < TILE_VOX) {
      int gb = tile * TILE_VOX + tid;
      if (gb >= B) gb = B - 1;
      sVidx[tid] = vox_indices[gb];
    }
    __syncthreads();
    if (tid < TILE_VOX * M_VIEWS) {
      int v = tid >> 3, m = tid & 7;
      int vi = sVidx[v];
      sCam[tid]  = camera_ids[vi * M_VIEWS + m];
      sMask[tid] = (scores[vi * M_VIEWS + m] < 0.f) ? -1e30f : 0.f;
    }
    if (tid < TILE_VOX * 3) {
      int v = tid / 3, a = tid % 3;
      float lo = (a == 0) ? lo0 : (a == 1) ? lo1 : lo2;
      float hi = (a == 0) ? hi0 : (a == 1) ? hi1 : hi2;
      float c  = voxels[(size_t)sVidx[v] * 3 + a];
      sXYZ[tid] = (c - lo) / (hi - lo) - 0.5f;
    }
    __syncthreads();

    // ---- Phase C: gather feats/extrinsics + positional encoding into A-frags ----
    for (int i = tid; i < AUGF_N; i += THREADS) {
      int mt = i >> 10; int ks = (i >> 9) & 1; int L = (i >> 4) & 31; int e = i & 15;
      int row = mt * 16 + (L & 15);
      int v = row >> 3, m = row & 7;
      int k = ks * 32 + ((e >> 3) << 4) + ((L >> 4) << 3) + (e & 7);
      float val = 0.f;
      if (k < F_DIM)
        val = voxel_features[((size_t)sVidx[v] * M_VIEWS + m) * F_DIM + k];
      else if (k < AUG_DIM)
        val = extrinsics[(size_t)sCam[v * M_VIEWS + m] * 16 + (k - F_DIM)];
      sAugF[i] = (f16)val;
    }
    for (int i = tid; i < ENCF_N; i += THREADS) {
      int qm = i >> 9; int L = (i >> 4) & 31; int e = i & 15;
      int mvox = qm * 16 + (L & 15);
      int k = ((e >> 3) << 4) + ((L >> 4) << 3) + (e & 7);
      float val = 0.f;
      if (k < SP_DIM) {
        int p = k / 3, a = k % 3;
        val = sinf(sXYZ[mvox * 3 + a] * 3.14159265358979f * (float)(1 << p));
      }
      sEncF[i] = (f16)val;
    }
    __syncthreads();

    // ---- Phase D: WMMA projections (no divergence: EXEC all ones) ----
    {
      // Q: wave = qm*8 + qn handles Q M-tile qm (16 voxels), N-tile qn
      int qm = wave >> 3, qn = wave & 7;
      v16h aq = *(const v16h*)(sEncF + (qm * 32 + lane) * 16);
      v16h bq = *(const v16h*)(sWqF + (qn * 32 + lane) * 16);
      v8f cq = {};
      cq = __builtin_amdgcn_wmma_f32_16x16x32_f16(false, aq, false, bq, (short)0, cq, false, false);
      {
        int rowb = qm * 16 + ((lane >> 4) << 3);
        int col  = qn * 16 + (lane & 15);
#pragma unroll
        for (int r = 0; r < 8; r++) sQ[(rowb + r) * HD + col] = cq[r];
      }

      // K/V: wave owns rows [16*wave, 16*wave+16) of the 256-row block
      v16h a0 = *(const v16h*)(sAugF + ((wave * 2 + 0) * 32 + lane) * 16); // K=0..31 (feats)
      v16h a1 = *(const v16h*)(sAugF + ((wave * 2 + 1) * 32 + lane) * 16); // K=32..63 (ext+pad)
#pragma unroll
      for (int nt = 0; nt < NT; nt++) {
        v16h bk0 = *(const v16h*)(sWkF + ((nt * 2 + 0) * 32 + lane) * 16);
        v16h bk1 = *(const v16h*)(sWkF + ((nt * 2 + 1) * 32 + lane) * 16);
        v8f ck = {};
        ck = __builtin_amdgcn_wmma_f32_16x16x32_f16(false, a0, false, bk0, (short)0, ck, false, false);
        ck = __builtin_amdgcn_wmma_f32_16x16x32_f16(false, a1, false, bk1, (short)0, ck, false, false);
        v16h bv = *(const v16h*)(sWvF + (nt * 32 + lane) * 16);
        v8f cv = {};
        cv = __builtin_amdgcn_wmma_f32_16x16x32_f16(false, a0, false, bv, (short)0, cv, false, false);
        int rowb = wave * 16 + ((lane >> 4) << 3);
        int col  = nt * 16 + (lane & 15);
#pragma unroll
        for (int r = 0; r < 8; r++) {
          sKh[(rowb + r) * HD + col] = (f16)ck[r];
          sVh[(rowb + r) * HD + col] = (f16)cv[r];
        }
      }
    }
    __syncthreads();

    // ---- Phase E: logits (16 threads per voxel, 2 logits each) ----
    {
      int vb = tid >> 4, r = tid & 15;
#pragma unroll
      for (int j = 0; j < 2; j++) {
        int l = r * 2 + j;          // 0..31 == h*8+m
        int h = l >> 3, m = l & 7;
        const float* qp = sQ + vb * HD + h * DIM_KEY;
        v16h k0 = *(const v16h*)(sKh + (vb * M_VIEWS + m) * HD + h * DIM_KEY);
        v16h k1 = *(const v16h*)(sKh + (vb * M_VIEWS + m) * HD + h * DIM_KEY + 16);
        float s = 0.f;
#pragma unroll
        for (int d = 0; d < 16; d++) s += qp[d] * (float)k0[d];
#pragma unroll
        for (int d = 0; d < 16; d++) s += qp[16 + d] * (float)k1[d];
        sLog[vb * 32 + l] = s * 0.17677669529663687f + sMask[vb * M_VIEWS + m];
      }
    }
    __syncthreads();

    // ---- Phase F: softmax + weighted V, store ----
    {
      int vb = tid >> 4, r = tid & 15;
      int h = r >> 2;               // one head per thread
      int dbase = (r & 3) * 8;      // 8 output dims within the head
      const float* lg = sLog + vb * 32 + h * M_VIEWS;
      float mx = lg[0];
#pragma unroll
      for (int m = 1; m < M_VIEWS; m++) mx = fmaxf(mx, lg[m]);
      float ex[M_VIEWS]; float sum = 0.f;
#pragma unroll
      for (int m = 0; m < M_VIEWS; m++) { ex[m] = __expf(lg[m] - mx); sum += ex[m]; }
      float inv = 1.f / sum;
      int gb = tile * TILE_VOX + vb;
      if (gb < B) {
        float acc[8];
#pragma unroll
        for (int j = 0; j < 8; j++) acc[j] = 0.f;
#pragma unroll
        for (int m = 0; m < M_VIEWS; m++) {
          v8h vv = *(const v8h*)(sVh + (vb * M_VIEWS + m) * HD + h * DIM_VALUE + dbase);
#pragma unroll
          for (int j = 0; j < 8; j++) acc[j] += ex[m] * (float)vv[j];
        }
#pragma unroll
        for (int j = 0; j < 8; j++)
          out[(size_t)gb * HD + h * DIM_VALUE + dbase + j] = acc[j] * inv;
      }
    }
  }
}

extern "C" void kernel_launch(void* const* d_in, const int* in_sizes, int n_in,
                              void* d_out, int out_size, void* d_ws, size_t ws_size,
                              hipStream_t stream) {
  const int*   vox_indices    = (const int*)d_in[0];
  const float* voxels         = (const float*)d_in[1];
  const float* voxel_features = (const float*)d_in[2];
  const float* scores         = (const float*)d_in[3];
  const int*   camera_ids     = (const int*)d_in[4];
  const float* extrinsics     = (const float*)d_in[5];
  const float* Wq             = (const float*)d_in[6];
  const float* Wk             = (const float*)d_in[7];
  const float* Wv             = (const float*)d_in[8];
  float* out  = (float*)d_out;
  unsigned* ws = (unsigned*)d_ws;

  const int B = in_sizes[0];
  const int N = in_sizes[1] / 3;

  init_minmax_kernel<<<1, 64, 0, stream>>>(ws);
  reduce_minmax_kernel<<<256, 256, 0, stream>>>(voxels, N, ws);

  (void)hipFuncSetAttribute(reinterpret_cast<const void*>(mvca_kernel),
                            hipFuncAttributeMaxDynamicSharedMemorySize,
                            (int)SMEM_BYTES);

  const int numTiles = (B + TILE_VOX - 1) / TILE_VOX;
  int grid = numTiles < 1536 ? numTiles : 1536;
  if (grid < 1) grid = 1;
  mvca_kernel<<<grid, THREADS, SMEM_BYTES, stream>>>(
      vox_indices, voxels, voxel_features, scores, camera_ids, extrinsics,
      Wq, Wk, Wv, out, ws, B, numTiles);
}
